// LSTMEncoder_45174466019919
// MI455X (gfx1250) — matrix operand order repaired and checked by reading, not compiled, not run on hardware
//
#include <hip/hip_runtime.h>
#include <hip/hip_bf16.h>

// ---------------------------------------------------------------------------
// MI455X (gfx1250) bidirectional LSTM encoder.
//   H=512, B=16, T=600.  bf16 WMMA (16x16x32, fp32 accum) for all GEMMs.
//   Recurrence: W_hh slices pinned in LDS (128KB/WG) across all 600 steps;
//   h state exchanged via L2 with a per-direction global arrival barrier.
//   LDS tile staging uses the Tensor Data Mover (TENSOR_LOAD_TO_LDS +
//   s_wait_tensorcnt) when the builtin is available.
// ---------------------------------------------------------------------------

#define Hn 512
#define Bn 16
#define Tn 600
#define G4 2048          // 4*H
#define LN_EPS 1e-5f

typedef __attribute__((ext_vector_type(16))) __bf16   v16bf;
typedef __attribute__((ext_vector_type(8)))  float    v8f;
typedef __attribute__((ext_vector_type(4)))  unsigned u32x4;

union Frag { v16bf v; u32x4 q[2]; };

// ---------------------------------------------------------------------------
// Tensor Data Mover (TDM) support.
//   ROCm 7.2 (clang-22): 5-arg builtin.  amdgpu-toolchain (clang-23, ships
//   amd_gfx1250_TDM.h): 6-arg builtin.  Fallback: manual LDS copy.
// ---------------------------------------------------------------------------
#if defined(__has_builtin)
#if __has_builtin(__builtin_amdgcn_tensor_load_to_lds) && \
    __has_builtin(__builtin_amdgcn_s_wait_tensorcnt)
#define USE_TDM 1
#if __has_include(<hip/amd_detail/amd_gfx1250_TDM.h>)
#define TDM_ARITY 6
#else
#define TDM_ARITY 5
#endif
#endif
#endif
#ifndef USE_TDM
#define USE_TDM 0
#endif

#if USE_TDM
typedef __attribute__((ext_vector_type(4))) unsigned tdm_u4;
typedef __attribute__((ext_vector_type(8))) int      tdm_i8;
typedef __attribute__((ext_vector_type(4))) int      tdm_i4;

// 2D tensor tile load (bf16 / data_size=2B), D# per CDNA5 ISA ch.8:
//  group0: [1:0]=count=1, [63:32]=lds byte addr, [120:64]=global addr, [127:126]=type=2
//  group1: data_size[17:16]=1, tensor_dim0[79:48], tensor_dim1[111:80],
//          tile_dim0[127:112], tile_dim1[143:128], tensor_dim0_stride[207:160]
__device__ __forceinline__ void tdm_load_bf16(const void* gsrc, void* ldst,
                                              unsigned dim0, unsigned dim1,
                                              unsigned tile0, unsigned tile1,
                                              unsigned stride0) {
  const unsigned long long ga = (unsigned long long)gsrc;
  const unsigned lo = (unsigned)(unsigned long long)ldst;  // low 32 bits = LDS byte offset
  tdm_u4 g0 = { 1u, lo, (unsigned)ga,
                ((unsigned)((ga >> 32) & 0x01FFFFFFull)) | 0x80000000u };
  tdm_i8 g1 = { (int)(1u << 16),                           // workgroup_mask=0, data_size=2B
                (int)((dim0 & 0xFFFFu) << 16),
                (int)((dim0 >> 16) | ((dim1 & 0xFFFFu) << 16)),
                (int)((dim1 >> 16) | ((tile0 & 0xFFFFu) << 16)),
                (int)(tile1 & 0xFFFFu),                    // tile_dim2 = 0
                (int)stride0, 0, 0 };                      // stride fits 32 bits here
  tdm_i4 gz = { 0, 0, 0, 0 };
#if TDM_ARITY == 6
  tdm_i8 gz8 = { 0, 0, 0, 0, 0, 0, 0, 0 };
  __builtin_amdgcn_tensor_load_to_lds(g0, g1, gz, gz, gz8, 0);
#else
  __builtin_amdgcn_tensor_load_to_lds(g0, g1, gz, gz, 0);
#endif
}
#define TDM_WAIT() __builtin_amdgcn_s_wait_tensorcnt((short)0)
#endif  // USE_TDM

// A fragment (16x32 bf16, MxK), ISA 7.12.2:
//  lanes 0-15 : M=lane,    K = {0..7} in q[0], {16..23} in q[1]
//  lanes 16-31: M=lane-16, K = {8..15} in q[0], {24..31} in q[1]
__device__ __forceinline__ v16bf load_afrag(const __bf16* base, int ld, int lane, int kc) {
  const int m = lane & 15, kh = lane >> 4;
  const __bf16* p = base + (size_t)m * ld + kc + kh * 8;
  Frag f;
  f.q[0] = *(const u32x4*)p;
  f.q[1] = *(const u32x4*)(p + 16);
  return f.v;
}

// B fragment (32x16 bf16, KxN) from row-major W[N][K]:
//  lanes 0-15 : N=lane,    K = {0..7} in q[0], {16..23} in q[1]
//  lanes 16-31: N=lane-16, K = {8..15} in q[0], {24..31} in q[1]
__device__ __forceinline__ v16bf load_bfrag(const __bf16* base, int ld, int lane, int kc, int n0) {
  const int n = lane & 15, kh = lane >> 4;
  const __bf16* p = base + (size_t)(n0 + n) * ld + kc + kh * 8;
  Frag f;
  f.q[0] = *(const u32x4*)p;
  f.q[1] = *(const u32x4*)(p + 16);
  return f.v;
}

#define WMMA_BF16(A, Bf, C) \
  __builtin_amdgcn_wmma_f32_16x16x32_bf16(false, (A), false, (Bf), (short)0, (C), false, false)

__device__ __forceinline__ v8f vzero8() { v8f z = {0.f,0.f,0.f,0.f,0.f,0.f,0.f,0.f}; return z; }
__device__ __forceinline__ float sigm(float x)  { return 1.0f / (1.0f + __expf(-x)); }
__device__ __forceinline__ float tanhfast(float x){ return 2.0f / (1.0f + __expf(-2.0f * x)) - 1.0f; }

// ---------------------------------------------------------------------------
// Prep kernels
// ---------------------------------------------------------------------------
__global__ void k_cvt_bf16(const float* __restrict__ s, __bf16* __restrict__ d, int n) {
  for (int i = blockIdx.x * blockDim.x + threadIdx.x; i < n; i += gridDim.x * blockDim.x)
    d[i] = (__bf16)s[i];
}

// x[B][H][T] fp32 -> xt[t*16+b][k] bf16  (rows of the WMMA A matrix = (t,b))
__global__ void k_transpose_x(const float* __restrict__ x, __bf16* __restrict__ xt) {
  int i = blockIdx.x * 256 + threadIdx.x;           // Tn*Bn*Hn total
  if (i >= Tn * Bn * Hn) return;
  int t = i / (Bn * Hn);
  int r = i - t * (Bn * Hn);
  int b = r >> 9, k = r & 511;
  xt[i] = (__bf16)x[((size_t)b * Hn + k) * Tn + t];
}

// zero h-state boundary slots (slot 0 and slot T+1) of both concat buffers + counters
__global__ void k_init(__bf16* __restrict__ h0, __bf16* __restrict__ h1, int* __restrict__ ctr) {
  int i = blockIdx.x * 256 + threadIdx.x;           // 4 * 16384 = 65536 total
  int buf = i >> 15, rest = i & 32767;
  int s = rest >> 14, e = rest & 16383;             // 16384 = 16*1024 elems per slot
  __bf16* p = buf ? h1 : h0;
  p[(s ? (size_t)(Tn + 1) * Bn * 1024 : 0) + e] = (__bf16)0.0f;
  if (i < 16) ctr[i] = 0;
}

// ---------------------------------------------------------------------------
// Parallel input GEMM: xp[dir][t][b][4H] = A[t,b,:] @ W[dir]^T + bias[dir]
// grid = (16, T), block = 256 (8 waves). Block covers 16 N-tiles of one dir.
// ---------------------------------------------------------------------------
__global__ void __launch_bounds__(256) k_gemm_xp(
    const __bf16* __restrict__ A, int Kin,
    const __bf16* __restrict__ W0, const __bf16* __restrict__ W1,
    const float* __restrict__ bias0, const float* __restrict__ bias1,
    float* __restrict__ xp0, float* __restrict__ xp1) {
  __shared__ __bf16 sA[16 * 1024];
  const int t = blockIdx.y;
  const int tid = threadIdx.x, lane = tid & 31, wave = tid >> 5;

  // stage the 16 x Kin A tile (rows (t,b) are contiguous -> 1D TDM tile)
#if USE_TDM
  if (tid < 32) {
    tdm_load_bf16(A + (size_t)t * 16 * Kin, sA,
                  16u * Kin, 1u, 16u * Kin, 1u, 16u * Kin);
    TDM_WAIT();
  }
#else
  {
    const u32x4* src = (const u32x4*)(A + (size_t)t * 16 * Kin);
    u32x4* dst = (u32x4*)sA;
    const int nch = 16 * Kin / 8;
    for (int i = tid; i < nch; i += 256) dst[i] = src[i];
  }
#endif
  __syncthreads();

  const int dir = (blockIdx.x >= 8);
  const __bf16* W   = dir ? W1 : W0;
  const float* bias = dir ? bias1 : bias0;
  float* xp         = dir ? xp1 : xp0;
  const int nt0  = (blockIdx.x & 7) * 16 + wave * 2;   // first of 2 N-tiles
  const int nc0  = nt0 * 16, nc1 = nc0 + 16;
  const int kh = lane >> 4, nn = lane & 15;

  v8f acc0 = vzero8(), acc1 = vzero8();
  #pragma unroll 4
  for (int kc = 0; kc < Kin; kc += 32) {
    v16bf a  = load_afrag(sA, Kin, lane, kc);
    v16bf b0 = load_bfrag(W, Kin, lane, kc, nc0);
    v16bf b1 = load_bfrag(W, Kin, lane, kc, nc1);
    acc0 = WMMA_BF16(a, b0, acc0);
    acc1 = WMMA_BF16(a, b1, acc1);
  }
  const float bv0 = bias[nc0 + nn], bv1 = bias[nc1 + nn];
  #pragma unroll
  for (int r = 0; r < 8; ++r) {
    const int m = r + kh * 8;
    xp[((size_t)t * 16 + m) * G4 + nc0 + nn] = acc0[r] + bv0;
    xp[((size_t)t * 16 + m) * G4 + nc1 + nn] = acc1[r] + bv1;
  }
}

// ---------------------------------------------------------------------------
// Recurrent scan. grid = 32 blocks (16/dir), block = 128 (4 waves).
// Each block owns 32 hidden units; its W_hh slice (4 gates x 32 units x 512)
// stays pinned in LDS (128 KB) for all 600 steps. Per step: TDM-stage prev h
// (16x512 bf16) from L2, 2 WMMA tiles per wave, gate math in fp32, write h
// slice, global arrival barrier per direction.
// hbuf layout: [T+2 slots][16 b][1024] bf16, time t -> slot t+1, slots 0 and
// T+1 are zeros (initial states for fwd/bwd).
// ---------------------------------------------------------------------------
__global__ void __launch_bounds__(128) k_lstm_rec(
    const float* __restrict__ xp0, const float* __restrict__ xp1,
    const __bf16* __restrict__ Whh0, const __bf16* __restrict__ Whh1,
    __bf16* __restrict__ hbuf, int* __restrict__ ctr) {
  __shared__ __bf16 sW[128 * 512];   // [gate*32 + unit][k]  (128 KB)
  __shared__ __bf16 sH[16 * 512];    // prev h, A-matrix layout (16 KB)
  __shared__ float  sG[8 * 16 * 16]; // [gate][unit-tile][m][n] accum (8 KB)

  const int tid = threadIdx.x, lane = tid & 31, wave = tid >> 5;
  const int dir = blockIdx.x >> 4, slice = blockIdx.x & 15;
  const int ubase = slice * 32;
  const __bf16* Whh = dir ? Whh1 : Whh0;
  const float* xp   = dir ? xp1  : xp0;
  int* myctr = ctr + dir;

  // pin W_hh slice in LDS: local row g*32+j  <-  global row g*512 + ubase + j
  // (rows within one gate block are contiguous -> 4 x 1D TDM tiles of 32KB)
#if USE_TDM
  if (tid < 32) {
    #pragma unroll
    for (int g = 0; g < 4; ++g)
      tdm_load_bf16(Whh + (size_t)(g * 512 + ubase) * 512, sW + g * 32 * 512,
                    32u * 512u, 1u, 32u * 512u, 1u, 32u * 512u);
    TDM_WAIT();
  }
#else
  for (int c = tid; c < 128 * 64; c += 128) {
    const int rrow = c >> 6, cc = c & 63;
    const int g = rrow >> 5, j = rrow & 31;
    ((u32x4*)sW)[c] = ((const u32x4*)(Whh + (size_t)(g * 512 + ubase + j) * 512))[cc];
  }
#endif

  const int ut = wave >> 1, gp = wave & 1;        // wave -> (unit-tile, gate pair)
  const int g0 = gp * 2, g1 = gp * 2 + 1;
  const int nb0 = g0 * 32 + ut * 16, nb1 = g1 * 32 + ut * 16;
  const int kh = lane >> 4, nn = lane & 15;

  const int eu = tid & 31, eb = tid >> 5;         // elementwise: unit, batch group
  const int eut = eu >> 4, en = eu & 15;
  float cst[4] = {0.f, 0.f, 0.f, 0.f};            // cell state in registers
  __syncthreads();

  for (int s = 0; s < Tn; ++s) {
    const int t = dir ? (Tn - 1 - s) : s;
    const int prevslot = dir ? (t + 2) : t;

    // stage previous h for this direction: 2D tile 16 rows x 512 elems,
    // global row stride 1024 elems (the fwd/bwd concat)
#if USE_TDM
    if (tid < 32) {
      tdm_load_bf16(hbuf + (size_t)prevslot * 16 * 1024 + dir * 512, sH,
                    512u, 16u, 512u, 16u, 1024u);
      TDM_WAIT();
    }
#else
    for (int c = tid; c < 1024; c += 128) {
      const int m = c >> 6, cc = c & 63;
      ((u32x4*)sH)[c] =
          ((const u32x4*)(hbuf + ((size_t)prevslot * 16 + m) * 1024 + dir * 512))[cc];
    }
#endif
    __syncthreads();

    v8f a0 = vzero8(), a1 = vzero8();
    #pragma unroll 4
    for (int kc = 0; kc < 512; kc += 32) {
      v16bf af = load_afrag(sH, 512, lane, kc);
      v16bf b0 = load_bfrag(sW, 512, lane, kc, nb0);
      v16bf b1 = load_bfrag(sW, 512, lane, kc, nb1);
      a0 = WMMA_BF16(af, b0, a0);
      a1 = WMMA_BF16(af, b1, a1);
    }
    #pragma unroll
    for (int r = 0; r < 8; ++r) {
      const int m = r + kh * 8;
      sG[((g0 * 2 + ut) * 16 + m) * 16 + nn] = a0[r];
      sG[((g1 * 2 + ut) * 16 + m) * 16 + nn] = a1[r];
    }
    __syncthreads();

    // gate math: each thread owns (unit eu, 4 batch rows); xp read once -> NT
    #pragma unroll
    for (int jj = 0; jj < 4; ++jj) {
      const int m = eb * 4 + jj;
      const size_t xr = ((size_t)t * 16 + m) * G4 + ubase + eu;
      const float pi = sG[((0 * 2 + eut) * 16 + m) * 16 + en] + __builtin_nontemporal_load(&xp[xr]);
      const float pf = sG[((1 * 2 + eut) * 16 + m) * 16 + en] + __builtin_nontemporal_load(&xp[xr + 512]);
      const float pg = sG[((2 * 2 + eut) * 16 + m) * 16 + en] + __builtin_nontemporal_load(&xp[xr + 1024]);
      const float po = sG[((3 * 2 + eut) * 16 + m) * 16 + en] + __builtin_nontemporal_load(&xp[xr + 1536]);
      const float cc = sigm(pf) * cst[jj] + sigm(pi) * tanhfast(pg);
      cst[jj] = cc;
      const float hh = sigm(po) * tanhfast(cc);
      hbuf[((size_t)(t + 1) * 16 + m) * 1024 + dir * 512 + ubase + eu] = (__bf16)hh;
    }
    if (s + 1 < Tn) {  // prefetch next step's xp row group
      const int tn = dir ? (t - 1) : (t + 1);
      __builtin_prefetch(&xp[((size_t)tn * 16 + eb * 4) * G4 + ubase + eu], 0, 1);
    }

    // per-direction global barrier (16 blocks), monotonic counter
    __threadfence();
    __syncthreads();
    if (tid == 0) {
      __hip_atomic_fetch_add(myctr, 1, __ATOMIC_RELEASE, __HIP_MEMORY_SCOPE_AGENT);
      while (__hip_atomic_load(myctr, __ATOMIC_ACQUIRE, __HIP_MEMORY_SCOPE_AGENT) < 16 * (s + 1))
        __builtin_amdgcn_s_sleep(1);
    }
    __syncthreads();
  }
}

// ---------------------------------------------------------------------------
// FC (WMMA) + LayerNorm + ReLU. grid = T, block = 256 (8 waves).
// ---------------------------------------------------------------------------
__global__ void __launch_bounds__(256) k_fc_ln(
    const __bf16* __restrict__ h1, const __bf16* __restrict__ fcw,
    const float* __restrict__ fcb, const float* __restrict__ lnw,
    const float* __restrict__ lnb, float* __restrict__ out) {
  __shared__ __bf16 sA[16 * 1024];
  __shared__ float  sO[16 * 512];
  __shared__ float  sRed[2][16][16];
  __shared__ float  sMV[2][16];
  const int t = blockIdx.x;
  const int tid = threadIdx.x, lane = tid & 31, wave = tid >> 5;

  // stage A = h1 concat for timestep t (slot t+1), contiguous 32KB -> 1D TDM
#if USE_TDM
  if (tid < 32) {
    tdm_load_bf16(h1 + (size_t)(t + 1) * 16 * 1024, sA,
                  16u * 1024u, 1u, 16u * 1024u, 1u, 16u * 1024u);
    TDM_WAIT();
  }
#else
  {
    const u32x4* src = (const u32x4*)(h1 + (size_t)(t + 1) * 16 * 1024);
    u32x4* dst = (u32x4*)sA;
    for (int i = tid; i < 2048; i += 256) dst[i] = src[i];
  }
#endif
  __syncthreads();

  const int kh = lane >> 4, nn = lane & 15;
  const int nbase = wave * 64;                     // 4 N-tiles per wave
  v8f acc[4] = {vzero8(), vzero8(), vzero8(), vzero8()};
  #pragma unroll 2
  for (int kc = 0; kc < 1024; kc += 32) {
    v16bf a = load_afrag(sA, 1024, lane, kc);
    #pragma unroll
    for (int j = 0; j < 4; ++j) {
      v16bf b = load_bfrag(fcw, 1024, lane, kc, nbase + j * 16);
      acc[j] = WMMA_BF16(a, b, acc[j]);
    }
  }
  #pragma unroll
  for (int j = 0; j < 4; ++j) {
    const int col = nbase + j * 16 + nn;
    const float bv = fcb[col];
    #pragma unroll
    for (int r = 0; r < 8; ++r)
      sO[(r + kh * 8) * 512 + col] = acc[j][r] + bv;
  }
  __syncthreads();

  // layernorm over 512 per (b) row
  const int row = tid >> 4, seg = tid & 15;
  float s1 = 0.f, s2 = 0.f;
  #pragma unroll
  for (int i = 0; i < 32; ++i) {
    const float v = sO[row * 512 + seg * 32 + i];
    s1 += v; s2 += v * v;
  }
  sRed[0][row][seg] = s1; sRed[1][row][seg] = s2;
  __syncthreads();
  if (tid < 16) {
    float a1 = 0.f, a2 = 0.f;
    #pragma unroll
    for (int i = 0; i < 16; ++i) { a1 += sRed[0][tid][i]; a2 += sRed[1][tid][i]; }
    const float mu = a1 / 512.0f;
    sMV[0][tid] = mu;
    sMV[1][tid] = rsqrtf(a2 / 512.0f - mu * mu + LN_EPS);
  }
  __syncthreads();
  const float mu = sMV[0][row], ir = sMV[1][row];
  #pragma unroll 4
  for (int i = 0; i < 32; ++i) {
    const int col = seg * 32 + i;
    const float v = (sO[row * 512 + col] - mu) * ir * lnw[col] + lnb[col];
    __builtin_nontemporal_store(fmaxf(v, 0.0f),
                                &out[((size_t)row * Tn + t) * 512 + col]);  // row == b
  }
}

// ---------------------------------------------------------------------------
extern "C" void kernel_launch(void* const* d_in, const int* in_sizes, int n_in,
                              void* d_out, int out_size, void* d_ws, size_t ws_size,
                              hipStream_t stream) {
  (void)in_sizes; (void)n_in; (void)out_size; (void)ws_size;
  const float* x     = (const float*)d_in[0];
  const float* wih0f = (const float*)d_in[1];
  const float* whh0f = (const float*)d_in[2];
  const float* b0f   = (const float*)d_in[3];
  const float* wih0b = (const float*)d_in[4];
  const float* whh0b = (const float*)d_in[5];
  const float* b0b   = (const float*)d_in[6];
  const float* wih1f = (const float*)d_in[7];
  const float* whh1f = (const float*)d_in[8];
  const float* b1f   = (const float*)d_in[9];
  const float* wih1b = (const float*)d_in[10];
  const float* whh1b = (const float*)d_in[11];
  const float* b1b   = (const float*)d_in[12];
  const float* fcw   = (const float*)d_in[13];
  const float* fcb   = (const float*)d_in[14];
  const float* lnw   = (const float*)d_in[15];
  const float* lnb   = (const float*)d_in[16];

  char* ws = (char*)d_ws;
  size_t off = 0;
  auto alloc = [&](size_t bytes) -> void* {
    void* p = ws + off;
    off = (off + bytes + 255) & ~(size_t)255;
    return p;
  };
  __bf16* xt  = (__bf16*)alloc((size_t)Tn * Bn * Hn * 2);
  __bf16* h0  = (__bf16*)alloc((size_t)(Tn + 2) * Bn * 1024 * 2);
  __bf16* h1  = (__bf16*)alloc((size_t)(Tn + 2) * Bn * 1024 * 2);
  float*  xp0 = (float*)alloc((size_t)Tn * Bn * G4 * 4);
  float*  xp1 = (float*)alloc((size_t)Tn * Bn * G4 * 4);
  __bf16* wih0f_b = (__bf16*)alloc((size_t)G4 * Hn * 2);
  __bf16* wih0b_b = (__bf16*)alloc((size_t)G4 * Hn * 2);
  __bf16* whh0f_b = (__bf16*)alloc((size_t)G4 * Hn * 2);
  __bf16* whh0b_b = (__bf16*)alloc((size_t)G4 * Hn * 2);
  __bf16* wih1f_b = (__bf16*)alloc((size_t)G4 * 2 * Hn * 2);
  __bf16* wih1b_b = (__bf16*)alloc((size_t)G4 * 2 * Hn * 2);
  __bf16* whh1f_b = (__bf16*)alloc((size_t)G4 * Hn * 2);
  __bf16* whh1b_b = (__bf16*)alloc((size_t)G4 * Hn * 2);
  __bf16* fcw_b   = (__bf16*)alloc((size_t)Hn * 2 * Hn * 2);
  int* ctr = (int*)alloc(64);

  // weight conversions fp32 -> bf16 (row-major [N][K] already WMMA-B friendly)
  k_cvt_bf16<<<2048, 256, 0, stream>>>(wih0f, wih0f_b, G4 * Hn);
  k_cvt_bf16<<<2048, 256, 0, stream>>>(wih0b, wih0b_b, G4 * Hn);
  k_cvt_bf16<<<2048, 256, 0, stream>>>(whh0f, whh0f_b, G4 * Hn);
  k_cvt_bf16<<<2048, 256, 0, stream>>>(whh0b, whh0b_b, G4 * Hn);
  k_cvt_bf16<<<2048, 256, 0, stream>>>(wih1f, wih1f_b, G4 * 2 * Hn);
  k_cvt_bf16<<<2048, 256, 0, stream>>>(wih1b, wih1b_b, G4 * 2 * Hn);
  k_cvt_bf16<<<2048, 256, 0, stream>>>(whh1f, whh1f_b, G4 * Hn);
  k_cvt_bf16<<<2048, 256, 0, stream>>>(whh1b, whh1b_b, G4 * Hn);
  k_cvt_bf16<<<2048, 256, 0, stream>>>(fcw, fcw_b, Hn * 2 * Hn);
  k_transpose_x<<<(Tn * Bn * Hn + 255) / 256, 256, 0, stream>>>(x, xt);
  k_init<<<256, 256, 0, stream>>>(h0, h1, ctr);

  // layer 0
  k_gemm_xp<<<dim3(16, Tn), 256, 0, stream>>>(xt, Hn, wih0f_b, wih0b_b, b0f, b0b, xp0, xp1);
  k_lstm_rec<<<32, 128, 0, stream>>>(xp0, xp1, whh0f_b, whh0b_b, h0, ctr);
  // layer 1 (A rows = h0 concat at slot t+1)
  k_gemm_xp<<<dim3(16, Tn), 256, 0, stream>>>(h0 + 16 * 1024, 2 * Hn,
                                              wih1f_b, wih1b_b, b1f, b1b, xp0, xp1);
  k_lstm_rec<<<32, 128, 0, stream>>>(xp0, xp1, whh1f_b, whh1b_b, h1, ctr + 2);
  // FC + LN + ReLU
  k_fc_ln<<<Tn, 256, 0, stream>>>(h1, fcw_b, fcb, lnw, lnb, (float*)d_out);
}
